// TinySelfAttention_14559939133536
// MI455X (gfx1250) — matrix operand order repaired and checked by reading, not compiled, not run on hardware
//
#include <hip/hip_runtime.h>

// ---------------------------------------------------------------------------
// TinySelfAttention (LoRA q/v) for MI455X gfx1250 — bf16 WMMA pipeline.
// B=2, T=2048, D=1024, H=16, DH=64, R=8, scaling=2.0
// ---------------------------------------------------------------------------

#define BB 2
#define TT 2048
#define DD 1024
#define HH 16
#define DH 64
#define RR 8

typedef __attribute__((ext_vector_type(16))) __bf16 v16bf;
typedef __attribute__((ext_vector_type(8)))  float  v8f;
typedef __attribute__((ext_vector_type(4)))  unsigned int u32x4;
typedef __attribute__((ext_vector_type(4)))  int v4i;

#if __has_builtin(__builtin_amdgcn_global_load_async_to_lds_b128) && \
    __has_builtin(__builtin_amdgcn_s_wait_asynccnt)
#define USE_ASYNC_LDS 1
typedef __attribute__((address_space(1))) v4i g_v4i;  // global
typedef __attribute__((address_space(3))) v4i l_v4i;  // LDS
#else
#define USE_ASYNC_LDS 0
#endif

// Load a 16-element bf16 fragment as two 16-byte vector loads.
static __device__ __forceinline__ v16bf ld16(const __bf16* p0, const __bf16* p1) {
  union { u32x4 u[2]; v16bf v; } t;
  t.u[0] = *reinterpret_cast<const u32x4*>(p0);
  t.u[1] = *reinterpret_cast<const u32x4*>(p1);
  return t.v;
}

static __device__ __forceinline__ v8f wmma_bf16(v16bf a, v16bf b, v8f c) {
  // D = A(16x32) * B(32x16) + C, f32 accumulate
  return __builtin_amdgcn_wmma_f32_16x16x32_bf16(
      /*neg_a=*/false, a, /*neg_b=*/false, b,
      /*c_mod=*/(short)0, c, /*reuse_a=*/false, /*reuse_b=*/false);
}

// ---------------------------------------------------------------------------
// f32 -> bf16 conversion (HBM-bound, trivial)
// ---------------------------------------------------------------------------
__global__ void f32_to_bf16(const float* __restrict__ in, __bf16* __restrict__ out, int n) {
  int i = blockIdx.x * blockDim.x + threadIdx.x;
  int stride = gridDim.x * blockDim.x;
  for (; i < n; i += stride) out[i] = (__bf16)in[i];
}

// ---------------------------------------------------------------------------
// LoRA first stage: xa[m, r] = sum_k x[m,k] * A[r,k]   (A is [R=8, D])
// ---------------------------------------------------------------------------
__global__ __launch_bounds__(32)
void lora_xa_kernel(const float* __restrict__ x, const float* __restrict__ A,
                    float* __restrict__ xa) {
  const int row  = blockIdx.x;
  const int lane = threadIdx.x;
  float s[RR];
#pragma unroll
  for (int r = 0; r < RR; ++r) s[r] = 0.f;
  const float* xr = x + (size_t)row * DD;
  for (int k = lane; k < DD; k += 32) {
    float xv = xr[k];
#pragma unroll
    for (int r = 0; r < RR; ++r) s[r] += xv * A[r * DD + k];
  }
#pragma unroll
  for (int r = 0; r < RR; ++r) {
#pragma unroll
    for (int off = 16; off >= 1; off >>= 1) s[r] += __shfl_xor(s[r], off, 32);
  }
  if (lane == 0) {
#pragma unroll
    for (int r = 0; r < RR; ++r) xa[(size_t)row * RR + r] = s[r];
  }
}

// ---------------------------------------------------------------------------
// WMMA GEMM:  O[M,N] = X[M,K] @ W[N,K]^T  (+ scaling * xa @ loraB^T)
// Block = 128 threads = 4 waves; wave computes a 32x64 output tile
// (2 A-frags x 4 B-tiles = 8 WMMAs per 32-deep k-step -> 8 independent
// accumulation chains to hide VMEM latency). grid = (M/32, N/256).
// ---------------------------------------------------------------------------
__global__ __launch_bounds__(128)
void proj_gemm(const __bf16* __restrict__ X, const __bf16* __restrict__ W,
               __bf16* __restrict__ Ob, float* __restrict__ Of,
               const float* __restrict__ xa, const float* __restrict__ loraB,
               float scaling, int M, int N, int K) {
  const int lane = threadIdx.x & 31;
  const int wave = threadIdx.x >> 5;
  const int row0 = blockIdx.x * 32;
  const int n0   = blockIdx.y * 256 + wave * 64;
  const int mrow = lane & 15;
  const int kbA  = (lane < 16) ? 0 : 8;    // A-frag K base (16-bit layout)
  const int kbB  = (lane < 16) ? 0 : 8;    // B-frag K base

  v8f acc[2][4] = {};
  const __bf16* xrow0 = X + (size_t)(row0 + mrow) * K;
  const __bf16* xrow1 = X + (size_t)(row0 + 16 + mrow) * K;

  for (int k0 = 0; k0 < K; k0 += 32) {
    v16bf a0 = ld16(xrow0 + k0 + kbA, xrow0 + k0 + kbA + 16);
    v16bf a1 = ld16(xrow1 + k0 + kbA, xrow1 + k0 + kbA + 16);
    if (k0 + 32 < K) {
      __builtin_prefetch(xrow0 + k0 + 32 + kbA, 0, 3);
      __builtin_prefetch(xrow1 + k0 + 32 + kbA, 0, 3);
    }
#pragma unroll
    for (int t = 0; t < 4; ++t) {
      const __bf16* wrow = W + (size_t)(n0 + t * 16 + mrow) * K + k0 + kbB;
      v16bf bfrag = ld16(wrow, wrow + 16);
      acc[0][t] = wmma_bf16(a0, bfrag, acc[0][t]);
      acc[1][t] = wmma_bf16(a1, bfrag, acc[1][t]);
    }
  }

  const int rloc = (lane < 16) ? 0 : 8;
#pragma unroll
  for (int rb = 0; rb < 2; ++rb) {
#pragma unroll
    for (int t = 0; t < 4; ++t) {
      const int col = n0 + t * 16 + mrow;
#pragma unroll
      for (int v = 0; v < 8; ++v) {
        const int row = row0 + rb * 16 + rloc + v;
        float val = acc[rb][t][v];
        if (loraB) {  // fused rank-8 LoRA delta
          const float* xr = xa + (size_t)row * RR;
          const float* lb = loraB + (size_t)col * RR;
          float d = 0.f;
#pragma unroll
          for (int r = 0; r < RR; ++r) d += xr[r] * lb[r];
          val += scaling * d;
        }
        if (Of) Of[(size_t)row * N + col] = val;
        else    Ob[(size_t)row * N + col] = (__bf16)val;
      }
    }
  }
}

// ---------------------------------------------------------------------------
// Flash attention, causal. Block = 256 threads = 8 waves; each wave owns a
// 16-row Q tile (block covers 128 rows of one (b,h)). Uniform 32-key loop:
// all waves share LDS-staged K (async-to-LDS when available) and V
// (transposed). Scores and PV via bf16 WMMA; online softmax via 16-lane
// shfl_xor reductions over the wave32 half-groups.
// ---------------------------------------------------------------------------
__global__ __launch_bounds__(256)
void flash_attn(const __bf16* __restrict__ Q, const __bf16* __restrict__ K,
                const __bf16* __restrict__ V, __bf16* __restrict__ O) {
  __shared__ alignas(16) __bf16 sK[32][DH];    // 32 keys x 64 dh
  __shared__ alignas(16) __bf16 sVT[DH][32];   // transposed: dh x keys
  __shared__ alignas(16) __bf16 sP[8][16][32]; // per-wave P restage (C->A layout)

  const int lane = threadIdx.x & 31;
  const int wave = threadIdx.x >> 5;
  const int tid  = threadIdx.x;
  const int bh   = blockIdx.y;
  const int b    = bh >> 4;
  const int h    = bh & 15;
  const int q0b  = blockIdx.x * 128;
  const int q0   = q0b + wave * 16;
  const int mrow = lane & 15;
  const int kbA  = (lane < 16) ? 0 : 8;
  const int kbB  = (lane < 16) ? 0 : 8;
  const int rloc = (lane < 16) ? 0 : 8;

  // Persistent Q fragments: 16x32 for dh-chunks [0,32) and [32,64)
  const __bf16* qbase = Q + (size_t)(b * TT + q0 + mrow) * DD + h * DH;
  v16bf aq0 = ld16(qbase + kbA,      qbase + kbA + 16);
  v16bf aq1 = ld16(qbase + 32 + kbA, qbase + 32 + kbA + 16);

  float mrw[8], lrw[8];
  v8f o[4] = {};
#pragma unroll
  for (int v = 0; v < 8; ++v) { mrw[v] = -1e30f; lrw[v] = 0.f; }

  const int kend = q0b + 127;  // uniform across block (causal upper bound)
  for (int kt0 = 0; kt0 <= kend; kt0 += 32) {
    // --- cooperative stage of 32x64 K and V tiles (V transposed) ---
    {
      int e = tid * 8, r = e >> 6, c = e & 63;
      const size_t grow = (size_t)(b * TT + kt0 + r) * DD + h * DH + c;
#if USE_ASYNC_LDS
      // CDNA5 async DMA: global -> LDS without touching VGPRs (ASYNCcnt)
      __builtin_amdgcn_global_load_async_to_lds_b128(
          (g_v4i*)(K + grow), (l_v4i*)&sK[r][c],
          /*offset=*/0, /*cpol=*/0);
#else
      *reinterpret_cast<u32x4*>(&sK[r][c]) =
          *reinterpret_cast<const u32x4*>(K + grow);
#endif
      union { u32x4 u; __bf16 e8[8]; } vv;
      vv.u = *reinterpret_cast<const u32x4*>(V + grow);
#pragma unroll
      for (int j = 0; j < 8; ++j) sVT[c + j][r] = vv.e8[j];
#if USE_ASYNC_LDS
      __builtin_amdgcn_s_wait_asynccnt(0);
#endif
    }
    __syncthreads();

    // --- scores: two 16x16 tiles over this 32-key block ---
    v8f s[2];
#pragma unroll
    for (int t = 0; t < 2; ++t) {
      v8f z = {};
      const __bf16* kp0 = &sK[t * 16 + mrow][kbB];
      z = wmma_bf16(aq0, ld16(kp0, kp0 + 16), z);
      const __bf16* kp1 = &sK[t * 16 + mrow][32 + kbB];
      z = wmma_bf16(aq1, ld16(kp1, kp1 + 16), z);
      s[t] = z;
    }

    // --- causal mask, scale, online softmax, restage P as bf16 ---
#pragma unroll
    for (int v = 0; v < 8; ++v) {
      const int rowg = q0 + rloc + v;
      const int c0 = kt0 + mrow, c1 = kt0 + 16 + mrow;
      float s0 = (c0 <= rowg) ? s[0][v] * 0.125f : -1e30f;
      float s1 = (c1 <= rowg) ? s[1][v] * 0.125f : -1e30f;
      float mx = fmaxf(s0, s1);
#pragma unroll
      for (int off = 8; off >= 1; off >>= 1) mx = fmaxf(mx, __shfl_xor(mx, off, 32));
      const float mn   = fmaxf(mrw[v], mx);
      const float corr = __expf(mrw[v] - mn);
      const float p0 = __expf(s0 - mn);
      const float p1 = __expf(s1 - mn);
      float rs = p0 + p1;
#pragma unroll
      for (int off = 8; off >= 1; off >>= 1) rs += __shfl_xor(rs, off, 32);
      lrw[v] = lrw[v] * corr + rs;
      mrw[v] = mn;
#pragma unroll
      for (int t2 = 0; t2 < 4; ++t2) o[t2][v] *= corr;
      sP[wave][rloc + v][mrow]      = (__bf16)p0;
      sP[wave][rloc + v][16 + mrow] = (__bf16)p1;
    }
    __syncthreads();

    // --- PV: o(16x64) += P(16x32) @ V(32x64) ---
    const __bf16* pp = &sP[wave][mrow][kbA];
    v16bf ap = ld16(pp, pp + 16);
#pragma unroll
    for (int d = 0; d < 4; ++d) {
      const __bf16* vp = &sVT[d * 16 + mrow][kbB];
      o[d] = wmma_bf16(ap, ld16(vp, vp + 16), o[d]);
    }
    __syncthreads();
  }

  // --- epilogue: normalize and store bf16 ---
#pragma unroll
  for (int d = 0; d < 4; ++d) {
#pragma unroll
    for (int v = 0; v < 8; ++v) {
      const int rowg = q0 + rloc + v;
      const float val = o[d][v] / lrw[v];
      O[(size_t)(b * TT + rowg) * DD + h * DH + d * 16 + mrow] = (__bf16)val;
    }
  }
}

// ---------------------------------------------------------------------------
// Host-side orchestration
// ---------------------------------------------------------------------------
extern "C" void kernel_launch(void* const* d_in, const int* in_sizes, int n_in,
                              void* d_out, int out_size, void* d_ws, size_t ws_size,
                              hipStream_t stream) {
  (void)in_sizes; (void)n_in; (void)out_size; (void)ws_size;
  const float* x    = (const float*)d_in[0];
  // d_in[1] = attn_mask (causal, recomputed analytically)
  const float* wq_w = (const float*)d_in[2];
  const float* wq_A = (const float*)d_in[3];
  const float* wq_B = (const float*)d_in[4];
  const float* wk_w = (const float*)d_in[5];
  const float* wv_w = (const float*)d_in[6];
  const float* wv_A = (const float*)d_in[7];
  const float* wv_B = (const float*)d_in[8];
  const float* wo_w = (const float*)d_in[9];
  float* out = (float*)d_out;

  const int M = BB * TT;        // 4096
  const size_t XW_BYTES = (size_t)M * DD * 2;       // 8 MB (bf16 activation)
  const size_t W_BYTES  = (size_t)DD * DD * 2;      // 2 MB per weight

  char* ws = (char*)d_ws;
  size_t off = 0;
  auto take = [&](size_t bytes) { char* p = ws + off; off += (bytes + 255) & ~(size_t)255; return p; };
  __bf16* xb  = (__bf16*)take(XW_BYTES);
  __bf16* wqb = (__bf16*)take(W_BYTES);
  __bf16* wkb = (__bf16*)take(W_BYTES);
  __bf16* wvb = (__bf16*)take(W_BYTES);
  __bf16* wob = (__bf16*)take(W_BYTES);
  __bf16* qb  = (__bf16*)take(XW_BYTES);
  __bf16* kb  = (__bf16*)take(XW_BYTES);
  __bf16* vb  = (__bf16*)take(XW_BYTES);
  __bf16* aob = (__bf16*)take(XW_BYTES);
  float*  xaq = (float*)take((size_t)M * RR * 4);
  float*  xav = (float*)take((size_t)M * RR * 4);

  // 1) f32 -> bf16 converts
  f32_to_bf16<<<2048, 256, 0, stream>>>(x,    xb,  M * DD);
  f32_to_bf16<<<1024, 256, 0, stream>>>(wq_w, wqb, DD * DD);
  f32_to_bf16<<<1024, 256, 0, stream>>>(wk_w, wkb, DD * DD);
  f32_to_bf16<<<1024, 256, 0, stream>>>(wv_w, wvb, DD * DD);
  f32_to_bf16<<<1024, 256, 0, stream>>>(wo_w, wob, DD * DD);

  // 2) LoRA rank-8 first stage (f32)
  lora_xa_kernel<<<M, 32, 0, stream>>>(x, wq_A, xaq);
  lora_xa_kernel<<<M, 32, 0, stream>>>(x, wv_A, xav);

  // 3) Q/K/V projections (WMMA; LoRA delta fused for Q and V)
  dim3 pg(M / 32, DD / 256);
  proj_gemm<<<pg, 128, 0, stream>>>(xb, wqb, qb, nullptr, xaq, wq_B, 2.0f, M, DD, DD);
  proj_gemm<<<pg, 128, 0, stream>>>(xb, wkb, kb, nullptr, nullptr, nullptr, 0.f, M, DD, DD);
  proj_gemm<<<pg, 128, 0, stream>>>(xb, wvb, vb, nullptr, xav, wv_B, 2.0f, M, DD, DD);

  // 4) causal flash attention (WMMA QK^T and PV)
  dim3 ag(TT / 128, BB * HH);
  flash_attn<<<ag, 256, 0, stream>>>(qb, kb, vb, aob);

  // 5) output projection, f32 store to d_out
  proj_gemm<<<pg, 128, 0, stream>>>(aob, wob, nullptr, out, nullptr, nullptr, 0.f, M, DD, DD);
}
